// WriteHead_62809601736863
// MI455X (gfx1250) — matrix operand order repaired and checked by reading, not compiled, not run on hardware
//
#include <hip/hip_runtime.h>

#define FDIM 256
#define MDIM 32768
#define BDIM 32

typedef __attribute__((ext_vector_type(16))) __bf16 v16bf;
typedef __attribute__((ext_vector_type(8)))  float  v8f;

__device__ __forceinline__ float fast_tanh(float x) {
#if __has_builtin(__builtin_amdgcn_tanhf)
  return __builtin_amdgcn_tanhf(x);
#else
  // tanh(x) = (e^{2x}-1)/(e^{2x}+1); e^{2x} = exp2(x * 2*log2(e))
  float t = __builtin_amdgcn_exp2f(x * 2.8853900817779268f);
  return (t - 1.0f) * __builtin_amdgcn_rcpf(t + 1.0f);
#endif
}

// ---------------- K0: pre-swizzle W1m into WMMA B-operand bf16 hi/lo images ----------------
// Layout: flat[((jt*8 + kc)*32 + lane)*16 + e], where for B (32x16 bf16):
//   col  n = lane & 15
//   K    k = kc*32 + (lane>>4)*16 + e
// One-time 64K-element pass; both images (2 x 128 KB) stay L2-resident.
__global__ __launch_bounds__(256)
void pack_b_kernel(const float* __restrict__ W1, __bf16* __restrict__ Bhi,
                   __bf16* __restrict__ Blo) {
  int idx  = blockIdx.x * 256 + threadIdx.x;   // 0..65535
  int e    = idx & 15;
  int lane = (idx >> 4) & 31;
  int kc   = (idx >> 9) & 7;
  int jt   = idx >> 12;
  int j = jt * 16 + (lane & 15);
  int k = kc * 32 + (lane >> 4) * 16 + e;
  float v = W1[j * (2 * FDIM) + FDIM + k];     // W1m half
  __bf16 h = (__bf16)v;
  Bhi[idx] = h;
  Blo[idx] = (__bf16)(v - (float)h);
}

// ---------------- K1: hxb[b][j] = b1[j] + sum_k input[b,k] * W1[j,k] ----------------
__global__ __launch_bounds__(256)
void hxb_kernel(const float* __restrict__ input, const float* __restrict__ W1,
                const float* __restrict__ b1, float* __restrict__ hxb) {
  int b = blockIdx.x;
  int j = threadIdx.x;
  const float* xin = input + b * FDIM;
  const float* wr  = W1 + j * (2 * FDIM);
  float acc = b1[j];
  for (int k = 0; k < FDIM; k += 4) {
    float4 x4 = *(const float4*)(xin + k);
    float4 w4 = *(const float4*)(wr + k);
    acc += x4.x * w4.x + x4.y * w4.y + x4.z * w4.z + x4.w * w4.w;
  }
  hxb[b * FDIM + j] = acc;
}

// ---------------- K2: fused WMMA hm-GEMM + tanh score reduction ----------------
// Block = 256 threads (8 waves), handles 32 memory rows.
// Phase 1: hm[0..31][0..255] = memory_tile @ W1m.T via bf16x3-split WMMA -> LDS.
//          B operands come pre-swizzled from pack_b_kernel (pure b128 loads).
// Phase 2: scores[b][m] = sum_j W2[j]*tanh(hxb[b][j] + hm[m][j]).
__global__ __launch_bounds__(256)
void score_kernel(const float* __restrict__ memA, const __bf16* __restrict__ Bhi,
                  const __bf16* __restrict__ Blo, const float* __restrict__ W2,
                  const float* __restrict__ hxb, float* __restrict__ scores) {
  __shared__ float hm_lds[32 * FDIM];     // 32 KB, hm tile
  __shared__ float hx_t[FDIM * BDIM];     // 32 KB, hx transposed [j][b] (bank-conflict free)

  const int t    = threadIdx.x;
  const int lane = t & 31;
  const int w    = t >> 5;
  const int n    = lane & 15;             // row (A) within 16
  const int half = lane >> 4;
  const int m_base  = blockIdx.x * 32;
  const int mt      = w >> 2;             // which 16-row m-tile (0/1)
  const int jt_base = (w & 3) * 4;        // 4 j-tiles per wave

  // Stage hx transposed into LDS
  for (int i = 0; i < 32; ++i) {
    int idx = t + 256 * i;                // 0..8191
    int b = idx >> 8, j = idx & 255;
    hx_t[j * BDIM + b] = hxb[idx];
  }

  // ---- WMMA phase (uniform control flow: EXEC all ones) ----
  v8f acc[4] = {};
  const float* arow = memA + (size_t)(m_base + mt * 16 + n) * FDIM;
  for (int kci = 0; kci < 8; ++kci) {
    const int kc = kci * 32;
    // A operand: 16x32 bf16. lane<16: K = kc+{0..15}; lane>=16: K = kc+{16..31}.
    const int ak = kc + half * 16;
    float4 a0 = *(const float4*)(arow + ak);
    float4 a1 = *(const float4*)(arow + ak + 4);
    float4 a2 = *(const float4*)(arow + ak + 8);
    float4 a3 = *(const float4*)(arow + ak + 12);
    float af[16] = {a0.x,a0.y,a0.z,a0.w, a1.x,a1.y,a1.z,a1.w,
                    a2.x,a2.y,a2.z,a2.w, a3.x,a3.y,a3.z,a3.w};
    v16bf ahi, alo;
#pragma unroll
    for (int e = 0; e < 16; ++e) {
      __bf16 h = (__bf16)af[e];
      ahi[e] = h;
      alo[e] = (__bf16)(af[e] - (float)h);
    }
#pragma unroll
    for (int jt4 = 0; jt4 < 4; ++jt4) {
      const int jt  = jt_base + jt4;
      const int off = (jt * 8 + kci) * 32 + lane;     // v16bf granules (32 B each)
      v16bf bhi = *(const v16bf*)(Bhi + (size_t)off * 16);
      v16bf blo = *(const v16bf*)(Blo + (size_t)off * 16);
      // bf16x3 split-precision accumulation (drop lo*lo term)
      acc[jt4] = __builtin_amdgcn_wmma_f32_16x16x32_bf16(false, ahi, false, bhi,
                                                         (short)0, acc[jt4], false, false);
      acc[jt4] = __builtin_amdgcn_wmma_f32_16x16x32_bf16(false, ahi, false, blo,
                                                         (short)0, acc[jt4], false, false);
      acc[jt4] = __builtin_amdgcn_wmma_f32_16x16x32_bf16(false, alo, false, bhi,
                                                         (short)0, acc[jt4], false, false);
    }
  }
  // Spill D tiles to LDS per C/D layout: VGPR r -> rows {r, r+8}, col = lane%16
#pragma unroll
  for (int jt4 = 0; jt4 < 4; ++jt4) {
    const int j0 = (jt_base + jt4) * 16;
#pragma unroll
    for (int r = 0; r < 8; ++r) {
      int ml = mt * 16 + half * 8 + r;
      hm_lds[ml * FDIM + j0 + n] = acc[jt4][r];
    }
  }
  __syncthreads();

  // ---- Score phase: 32 b x 32 m per block; lane = b, wave+i = m ----
  const int b = t & 31;
  const int wv = t >> 5;
  for (int i = 0; i < 4; ++i) {
    const int ml = wv + 8 * i;
    const float* hmrow = &hm_lds[ml * FDIM];
    float s = 0.0f;
    for (int j = 0; j < FDIM; j += 4) {
      float4 hm4 = *(const float4*)(hmrow + j);          // broadcast within wave
      float4 w4  = *(const float4*)(W2 + j);             // uniform -> scalar loads
      float hx0 = hx_t[(j + 0) * BDIM + b];
      float hx1 = hx_t[(j + 1) * BDIM + b];
      float hx2 = hx_t[(j + 2) * BDIM + b];
      float hx3 = hx_t[(j + 3) * BDIM + b];
      s += w4.x * fast_tanh(hx0 + hm4.x);
      s += w4.y * fast_tanh(hx1 + hm4.y);
      s += w4.z * fast_tanh(hx2 + hm4.z);
      s += w4.w * fast_tanh(hx3 + hm4.w);
    }
    scores[(size_t)b * MDIM + m_base + ml] = s;
  }
}

// ---------------- K3: per-batch argmax + softmax max-weight test ----------------
__global__ __launch_bounds__(256)
void reduce_kernel(const float* __restrict__ scores, const float* __restrict__ thr,
                   int* __restrict__ best, int* __restrict__ ok) {
  __shared__ float rmax[256];
  __shared__ int   ridx[256];
  __shared__ float rsum[256];
  __shared__ float s_smax;
  __shared__ int   s_sidx;
  const int b = blockIdx.x, t = threadIdx.x;
  const float* row = scores + (size_t)b * MDIM;
  float vmax = -3.4e38f;
  int   vidx = 0x7fffffff;
  for (int i = 0; i < MDIM / 256; ++i) {
    int m = t + 256 * i;
    float v = row[m];
    if (v > vmax || (v == vmax && m < vidx)) { vmax = v; vidx = m; }
  }
  rmax[t] = vmax; ridx[t] = vidx;
  __syncthreads();
  if (t == 0) {
    float sm = rmax[0]; int si = ridx[0];
    for (int i = 1; i < 256; ++i)
      if (rmax[i] > sm || (rmax[i] == sm && ridx[i] < si)) { sm = rmax[i]; si = ridx[i]; }
    s_smax = sm; s_sidx = si;
  }
  __syncthreads();
  const float sm = s_smax;
  float acc = 0.0f;
  for (int i = 0; i < MDIM / 256; ++i) {
    int m = t + 256 * i;
    acc += __builtin_amdgcn_exp2f((row[m] - sm) * 1.4426950408889634f);
  }
  rsum[t] = acc;
  __syncthreads();
  if (t == 0) {
    float s = 0.0f;
    for (int i = 0; i < 256; ++i) s += rsum[i];
    float maxw = 1.0f / s;            // exp(smax - smax)/sum
    best[b] = s_sidx;
    ok[b]   = (maxw > thr[0]) ? 1 : 0;
  }
}

// ---------------- K4: output = memory with winning rows replaced ----------------
__global__ __launch_bounds__(256)
void write_kernel(const float* __restrict__ memA, const float* __restrict__ input,
                  const int* __restrict__ best, const int* __restrict__ ok,
                  float* __restrict__ out) {
  const int t   = threadIdx.x;
  const int row = blockIdx.x * 4 + (t >> 6);
  const int col = (t & 63) * 4;
  int src = -1;                       // segment_max: largest b (order=b+1) wins
  for (int b = 0; b < BDIM; ++b)
    if (ok[b] && best[b] == row) src = b;
  const float* sp = (src >= 0) ? (input + (size_t)src * FDIM)
                               : (memA + (size_t)row * FDIM);
  *(float4*)(out + (size_t)row * FDIM + col) = *(const float4*)(sp + col);
}

extern "C" void kernel_launch(void* const* d_in, const int* in_sizes, int n_in,
                              void* d_out, int out_size, void* d_ws, size_t ws_size,
                              hipStream_t stream) {
  (void)in_sizes; (void)n_in; (void)out_size; (void)ws_size;
  const float* input  = (const float*)d_in[0];
  const float* memory = (const float*)d_in[1];
  const float* W1     = (const float*)d_in[2];
  const float* b1     = (const float*)d_in[3];
  const float* W2     = (const float*)d_in[4];
  // d_in[5] (b2) unused: softmax/argmax are shift-invariant.
  const float* thr    = (const float*)d_in[6];

  char*   ws     = (char*)d_ws;
  float*  hxb    = (float*)ws;                          // 32*256*4   = 32 KB
  float*  scores = (float*)(ws + 32768);                // 32*32768*4 = 4 MB
  int*    best   = (int*)(ws + 32768 + 4194304);        // 32 ints
  int*    ok     = (int*)(ws + 32768 + 4194304 + 128);  // 32 ints
  __bf16* Bhi    = (__bf16*)(ws + 32768 + 4194304 + 256);           // 128 KB
  __bf16* Blo    = (__bf16*)(ws + 32768 + 4194304 + 256 + 131072);  // 128 KB
  float*  out    = (float*)d_out;

  pack_b_kernel<<<FDIM,     256, 0, stream>>>(W1, Bhi, Blo);
  hxb_kernel   <<<BDIM,     256, 0, stream>>>(input, W1, b1, hxb);
  score_kernel <<<MDIM / 32, 256, 0, stream>>>(memory, Bhi, Blo, W2, hxb, scores);
  reduce_kernel<<<BDIM,     256, 0, stream>>>(scores, thr, best, ok);
  write_kernel <<<MDIM / 4,  256, 0, stream>>>(memory, input, best, ok, out);
}
